// MAMIL1D_26130581029414
// MI455X (gfx1250) — matrix operand order repaired and checked by reading, not compiled, not run on hardware
//
#include <hip/hip_runtime.h>
#include <hip/hip_bf16.h>
#include <math.h>

typedef unsigned int u32;
typedef unsigned short u16;

typedef __attribute__((ext_vector_type(8)))  float  v8f;
typedef __attribute__((ext_vector_type(16))) __bf16 v16bf;
typedef __attribute__((ext_vector_type(4)))  int    v4i;

union FragBF { v16bf v; uint4 u[2]; };

__device__ inline u16 f2bf(float f) {
    u32 u = __float_as_uint(f);
    u32 r = (u + 0x7FFFu + ((u >> 16) & 1u)) >> 16;
    return (u16)r;
}

__device__ inline v8f wmma_bf16(v16bf a, v16bf b, v8f c) {
    return __builtin_amdgcn_wmma_f32_16x16x32_bf16(
        /*neg_a=*/false, a, /*neg_b=*/false, b,
        /*c_mod=*/(short)0, c, /*reuse_a=*/false, /*reuse_b=*/false);
}

// ---- CDNA5 async global->LDS direct path (ASYNCcnt), with safe fallback ----
#if defined(__has_builtin)
#if __has_builtin(__builtin_amdgcn_global_load_async_to_lds_b128)
#define HAVE_ASYNC_LDS 1
#endif
#endif

__device__ inline void copy16_g2lds(const void* gsrc, void* lds_dst) {
#ifdef HAVE_ASYNC_LDS
    typedef v4i __attribute__((address_space(1)))* gptr_t;
    typedef v4i __attribute__((address_space(3)))* lptr_t;
    __builtin_amdgcn_global_load_async_to_lds_b128(
        (gptr_t)(unsigned long long)gsrc,
        (lptr_t)(unsigned int)(unsigned long long)lds_dst,
        /*offset=*/0, /*cpol=*/0);
#else
    *(uint4*)lds_dst = *(const uint4*)gsrc;
#endif
}

__device__ inline void async_lds_wait() {
#ifdef HAVE_ASYNC_LDS
#if __has_builtin(__builtin_amdgcn_s_wait_asynccnt)
    __builtin_amdgcn_s_wait_asynccnt(0);
#else
    asm volatile("s_wait_asynccnt 0x0" ::: "memory");
#endif
#endif
}

// ---------------------------------------------------------------- prep
__global__ void cast_bf16_kernel(const float* __restrict__ src,
                                 u16* __restrict__ dst, int n) {
    int i = blockIdx.x * 256 + threadIdx.x;
    if (i < n) dst[i] = f2bf(src[i]);
}

// conv2_w [50][500] -> bf16 padded [64][512]
__global__ void prep_conv2w_kernel(const float* __restrict__ w,
                                   u16* __restrict__ dst) {
    int i = blockIdx.x * 256 + threadIdx.x;
    if (i < 64 * 512) {
        int o = i >> 9, k = i & 511;
        float v = (o < 50 && k < 500) ? w[o * 500 + k] : 0.f;
        dst[i] = f2bf(v);
    }
}

// ---------------------------------------------------------------- conv1 + relu + pool
// x[n][28][28] -> pooled1[n][20][12][12]
__global__ __launch_bounds__(256)
void conv1_pool_kernel(const float* __restrict__ x,
                       const float* __restrict__ w,
                       const float* __restrict__ b,
                       float* __restrict__ pooled1) {
    __shared__ float sx[784];
    __shared__ float sw[500];
    __shared__ float sb[20];
    int n = blockIdx.x, tid = threadIdx.x;
    for (int i = tid; i < 784; i += 256) sx[i] = x[n * 784 + i];
    for (int i = tid; i < 500; i += 256) sw[i] = w[i];
    if (tid < 20) sb[tid] = b[tid];
    __syncthreads();
    for (int idx = tid; idx < 2880; idx += 256) {
        int c = idx / 144, rem = idx % 144;
        int y = rem / 12, xo = rem % 12;
        float m = -1e30f;
        for (int dy = 0; dy < 2; ++dy)
            for (int dx = 0; dx < 2; ++dx) {
                int oy = 2 * y + dy, ox = 2 * xo + dx;
                float s = sb[c];
                for (int kh = 0; kh < 5; ++kh)
                    for (int kw = 0; kw < 5; ++kw)
                        s += sx[(oy + kh) * 28 + ox + kw] * sw[c * 25 + kh * 5 + kw];
                m = fmaxf(m, s);
            }
        pooled1[n * 2880 + idx] = fmaxf(m, 0.f);
    }
}

// ---------------------------------------------------------------- conv2 implicit-GEMM WMMA + relu + pool
// rows = 2 instances x 64 pixels, cols = 64 (50 valid), K = 512 (500 valid)
__global__ __launch_bounds__(256)
void conv2_pool_wmma_kernel(const float* __restrict__ pooled1,
                            const u16* __restrict__ Wbf,   // [64][512]
                            const float* __restrict__ c2b,
                            float* __restrict__ pooled2f,  // [N][800]
                            u16* __restrict__ pooled2bf) {
    __shared__ __align__(16) u16 lds_a[128][32];
    __shared__ __align__(16) u16 lds_b[64][32];
    int tid = threadIdx.x, wave = tid >> 5, lane = tid & 31;
    int n0 = blockIdx.x * 2;
    v8f acc[4];
    for (int i = 0; i < 4; ++i)
        for (int j = 0; j < 8; ++j) acc[i][j] = 0.f;

    const char* Bb = (const char*)Wbf;
    char* lb = (char*)&lds_b[0][0];

    for (int ks = 0; ks < 512; ks += 32) {
        __syncthreads();
        // B tile: 64 rows x 64B -> 256 x 16B async chunks (one per thread)
        {
            int d = tid;                 // 0..255
            int nn = d >> 2, c16 = d & 3;
            copy16_g2lds(Bb + (nn * 512 + ks) * 2 + c16 * 16, lb + d * 16);
        }
        // im2col A tile (fp32 gather -> bf16), VALU path
        for (int e = tid; e < 128 * 32; e += 256) {
            int r = e >> 5, k = e & 31;
            int kg = ks + k;
            float val = 0.f;
            if (kg < 500) {
                int nl = r >> 6, p = r & 63;
                int py = p >> 3, px = p & 7;
                int ci = kg / 25, rem = kg - ci * 25;
                int kh = rem / 5, kw = rem - kh * 5;
                val = pooled1[(n0 + nl) * 2880 + ci * 144 + (py + kh) * 12 + (px + kw)];
            }
            lds_a[r][k] = f2bf(val);
        }
        async_lds_wait();
        __syncthreads();
        int M = lane & 15;
        int rr = wave * 16 + M;
        int k0 = (lane < 16) ? 0 : 8;
        FragBF a;
        a.u[0] = *(const uint4*)&lds_a[rr][k0];
        a.u[1] = *(const uint4*)&lds_a[rr][k0 + 16];
        int k0b = (lane < 16) ? 0 : 8;
        for (int ct = 0; ct < 4; ++ct) {
            FragBF bfrag;
            int nn = ct * 16 + M;
            bfrag.u[0] = *(const uint4*)&lds_b[nn][k0b];
            bfrag.u[1] = *(const uint4*)&lds_b[nn][k0b + 16];
            acc[ct] = wmma_bf16(a.v, bfrag.v, acc[ct]);
        }
    }
    // fused relu + 2x2 maxpool from C fragments:
    // wave strip: n_loc = wave/4, pool-row y = wave%4; M = px + 8*(py&1)
    int nl = wave >> 2, y = wave & 3;
    int lane15 = lane & 15;
    for (int ct = 0; ct < 4; ++ct) {
        int c = ct * 16 + lane15;
        for (int x = 0; x < 4; ++x) {
            float a0 = fmaxf(acc[ct][2 * x], acc[ct][2 * x + 1]);
            float a1 = __shfl_xor(a0, 16, 32);
            float mm = fmaxf(a0, a1);
            if (lane < 16 && c < 50) {
                float val = fmaxf(mm + c2b[c], 0.f);
                int off = (n0 + nl) * 800 + c * 16 + y * 4 + x;
                pooled2f[off] = val;
                pooled2bf[off] = f2bf(val);
            }
        }
    }
}

// ---------------------------------------------------------------- generic 128-row bf16 WMMA GEMM
// C[row][c] = epilogue(sum_k A[row][k]*B[c][k] + bias[c])
// MODE 0: relu, write f32 + bf16 ; MODE 1: tanh, write f32 only
template <int BN, int MODE>
__global__ __launch_bounds__(256)
void gemm_bf16_kernel(const u16* __restrict__ Abf,
                      const u16* __restrict__ Bbf,   // [BN][K]
                      const float* __restrict__ bias,
                      float* __restrict__ outF,
                      u16* __restrict__ outBF,
                      int K) {
    __shared__ __align__(16) u16 lds_a[128][32];
    __shared__ __align__(16) u16 lds_b[BN][32];
    int tid = threadIdx.x, wave = tid >> 5, lane = tid & 31;
    int row0 = blockIdx.x * 128;
    v8f acc[BN / 16];
    for (int i = 0; i < BN / 16; ++i)
        for (int j = 0; j < 8; ++j) acc[i][j] = 0.f;

    const char* Ab = (const char*)Abf;
    const char* Bb = (const char*)Bbf;
    char* la = (char*)&lds_a[0][0];
    char* lb = (char*)&lds_b[0][0];

    for (int ks = 0; ks < K; ks += 32) {
        __syncthreads();
        // A tile: 128 rows x 64B -> 512 x 16B async chunks
        for (int d = tid; d < 128 * 4; d += 256) {
            int r = d >> 2, c16 = d & 3;
            copy16_g2lds(Ab + ((size_t)(row0 + r) * K + ks) * 2 + c16 * 16, la + d * 16);
        }
        // B tile: BN rows x 64B -> BN*4 x 16B async chunks
        for (int d = tid; d < BN * 4; d += 256) {
            int nn = d >> 2, c16 = d & 3;
            copy16_g2lds(Bb + (nn * K + ks) * 2 + c16 * 16, lb + d * 16);
        }
        async_lds_wait();
        __syncthreads();
        int M = lane & 15;
        int rr = wave * 16 + M;
        int k0 = (lane < 16) ? 0 : 8;
        FragBF a;
        a.u[0] = *(const uint4*)&lds_a[rr][k0];
        a.u[1] = *(const uint4*)&lds_a[rr][k0 + 16];
        int k0b = (lane < 16) ? 0 : 8;
        for (int ct = 0; ct < BN / 16; ++ct) {
            FragBF bfrag;
            int nn = ct * 16 + M;
            bfrag.u[0] = *(const uint4*)&lds_b[nn][k0b];
            bfrag.u[1] = *(const uint4*)&lds_b[nn][k0b + 16];
            acc[ct] = wmma_bf16(a.v, bfrag.v, acc[ct]);
        }
    }
    for (int ct = 0; ct < BN / 16; ++ct) {
        int c = ct * 16 + (lane & 15);
        float bb = bias[c];
        for (int v = 0; v < 8; ++v) {
            int m = v + ((lane >= 16) ? 8 : 0);
            int row = row0 + wave * 16 + m;
            float val = acc[ct][v] + bb;
            if (MODE == 0) {
                val = fmaxf(val, 0.f);
                outF[row * BN + c] = val;
                outBF[row * BN + c] = f2bf(val);
            } else {
                val = tanhf(val);
                outF[row * BN + c] = val;
            }
        }
    }
}

// ---------------------------------------------------------------- neighbor attention (keys precomputed as Kmat)
__global__ void neighbor_kernel(const float* __restrict__ H,
                                const float* __restrict__ Km,
                                float* __restrict__ Hc,
                                u16* __restrict__ Hcbf) {
    int n = blockIdx.x * 256 + threadIdx.x;
    if (n >= 8192) return;
    int l = (n == 0) ? 1 : n - 1;
    int r = (n == 8191) ? 8190 : n + 1;
    float s0 = 0.f, s1 = 0.f;
    for (int d = 0; d < 128; ++d) {
        float h = H[n * 128 + d];
        s0 += Km[l * 128 + d] * h;
        s1 += Km[r * 128 + d] * h;
    }
    float m = fmaxf(s0, s1);
    float e0 = expf(s0 - m), e1 = expf(s1 - m);
    float inv = 1.f / (e0 + e1);
    float a0 = e0 * inv, a1 = e1 * inv;
    for (int d = 0; d < 128; ++d) {
        float h = H[n * 128 + d];
        float nb = a0 * H[l * 128 + d] + a1 * H[r * 128 + d];
        Hc[n * 256 + d] = h;        Hcbf[n * 256 + d] = f2bf(h);
        Hc[n * 256 + 128 + d] = nb; Hcbf[n * 256 + 128 + d] = f2bf(nb);
    }
}

// ---------------------------------------------------------------- template scores
__global__ void scores_kernel(const float* __restrict__ P,
                              const float* __restrict__ tmpl,
                              float* __restrict__ scores) {
    int n = blockIdx.x * 256 + threadIdx.x;
    if (n >= 8192) return;
    for (int t = 0; t < 10; ++t) {
        float s = 0.f;
        for (int c = 0; c < 256; ++c) s += P[n * 256 + c] * tmpl[t * 256 + c];
        scores[t * 8192 + n] = s;
    }
}

// ---------------------------------------------------------------- per-template softmax over N + weighted pooling
__global__ __launch_bounds__(256)
void softmax_embs_kernel(const float* __restrict__ scores,
                         const float* __restrict__ Hc,
                         float* __restrict__ betas,
                         float* __restrict__ embs) {
    __shared__ float red[256];
    __shared__ float s_max, s_sum;
    int t = blockIdx.x, tid = threadIdx.x;
    const float* sc = scores + t * 8192;
    float m = -1e30f;
    for (int n = tid; n < 8192; n += 256) m = fmaxf(m, sc[n]);
    red[tid] = m; __syncthreads();
    for (int s = 128; s > 0; s >>= 1) {
        if (tid < s) red[tid] = fmaxf(red[tid], red[tid + s]);
        __syncthreads();
    }
    if (tid == 0) s_max = red[0];
    __syncthreads();
    float mx = s_max;
    float sum = 0.f;
    for (int n = tid; n < 8192; n += 256) sum += expf(sc[n] - mx);
    red[tid] = sum; __syncthreads();
    for (int s = 128; s > 0; s >>= 1) {
        if (tid < s) red[tid] += red[tid + s];
        __syncthreads();
    }
    if (tid == 0) s_sum = red[0];
    __syncthreads();
    float inv = 1.f / s_sum;
    for (int n = tid; n < 8192; n += 256) betas[t * 8192 + n] = expf(sc[n] - mx) * inv;
    // embs[t][tid] = sum_n beta * Hc[n][tid]   (blockDim == 256 == L2)
    float acc = 0.f;
    for (int n = 0; n < 8192; ++n) acc += expf(sc[n] - mx) * inv * Hc[n * 256 + tid];
    embs[t * 256 + tid] = acc;
}

// ---------------------------------------------------------------- global attention head (1 block)
__global__ __launch_bounds__(256)
void global_attn_kernel(const float* __restrict__ embs,
                        const float* __restrict__ gw1, const float* __restrict__ gb1,
                        const float* __restrict__ gw2, const float* __restrict__ gb2,
                        const float* __restrict__ cw,  const float* __restrict__ cb,
                        float* __restrict__ gammas, float* __restrict__ out) {
    __shared__ float hg[10 * 128];
    __shared__ float g[10];
    __shared__ float gam[10];
    __shared__ float red[256];
    int tid = threadIdx.x;
    for (int idx = tid; idx < 1280; idx += 256) {
        int t = idx >> 7, d = idx & 127;
        float s = gb1[d];
        for (int k = 0; k < 256; ++k) s += embs[t * 256 + k] * gw1[d * 256 + k];
        hg[idx] = tanhf(s);
    }
    __syncthreads();
    if (tid < 10) {
        float s = gb2[0];
        for (int d = 0; d < 128; ++d) s += hg[tid * 128 + d] * gw2[d];
        g[tid] = s;
    }
    __syncthreads();
    if (tid == 0) {
        float m = -1e30f;
        for (int t = 0; t < 10; ++t) m = fmaxf(m, g[t]);
        float sum = 0.f;
        for (int t = 0; t < 10; ++t) { gam[t] = expf(g[t] - m); sum += gam[t]; }
        for (int t = 0; t < 10; ++t) { gam[t] /= sum; gammas[t] = gam[t]; }
    }
    __syncthreads();
    {
        float s = 0.f;
        for (int t = 0; t < 10; ++t) s += gam[t] * embs[t * 256 + tid];
        red[tid] = s * cw[tid];
    }
    __syncthreads();
    for (int s = 128; s > 0; s >>= 1) {
        if (tid < s) red[tid] += red[tid + s];
        __syncthreads();
    }
    if (tid == 0) {
        float logit = red[0] + cb[0];
        float yp = 1.f / (1.f + expf(-logit));
        out[0] = yp;
        out[1] = (yp >= 0.5f) ? 1.f : 0.f;
    }
}

// ---------------------------------------------------------------- A = gammas @ betas
__global__ void attnmap_kernel(const float* __restrict__ gammas,
                               const float* __restrict__ betas,
                               float* __restrict__ out) {
    int n = blockIdx.x * 256 + threadIdx.x;
    if (n >= 8192) return;
    float s = 0.f;
    for (int t = 0; t < 10; ++t) s += gammas[t] * betas[t * 8192 + n];
    out[2 + n] = s;
}

// ================================================================ launcher
extern "C" void kernel_launch(void* const* d_in, const int* in_sizes, int n_in,
                              void* d_out, int out_size, void* d_ws, size_t ws_size,
                              hipStream_t stream) {
    const int N = 8192;
    const float* x       = (const float*)d_in[0];
    const float* conv1_w = (const float*)d_in[1];
    const float* conv1_b = (const float*)d_in[2];
    const float* conv2_w = (const float*)d_in[3];
    const float* conv2_b = (const float*)d_in[4];
    const float* fe2_w   = (const float*)d_in[5];
    const float* fe2_b   = (const float*)d_in[6];
    const float* nb_w    = (const float*)d_in[7];
    const float* nb_b    = (const float*)d_in[8];
    const float* tmpl    = (const float*)d_in[9];
    const float* proto_w = (const float*)d_in[10];
    const float* proto_b = (const float*)d_in[11];
    const float* glob_w1 = (const float*)d_in[12];
    const float* glob_b1 = (const float*)d_in[13];
    const float* glob_w2 = (const float*)d_in[14];
    const float* glob_b2 = (const float*)d_in[15];
    const float* cls_w   = (const float*)d_in[16];
    const float* cls_b   = (const float*)d_in[17];
    float* out = (float*)d_out;

    char* ws = (char*)d_ws;
    size_t off = 0;
    auto alloc = [&](size_t bytes) -> void* {
        off = (off + 255) & ~(size_t)255;
        void* p = ws + off;
        off += bytes;
        return p;
    };
    float* pooled1   = (float*)alloc((size_t)N * 2880 * 4);
    u16*   conv2wbf  = (u16*)  alloc((size_t)64 * 512 * 2);
    float* pooled2f  = (float*)alloc((size_t)N * 800 * 4);
    u16*   pooled2bf = (u16*)  alloc((size_t)N * 800 * 2);
    u16*   fe2wbf    = (u16*)  alloc((size_t)128 * 800 * 2);
    u16*   nbwbf     = (u16*)  alloc((size_t)128 * 128 * 2);
    u16*   protowbf  = (u16*)  alloc((size_t)256 * 256 * 2);
    float* H         = (float*)alloc((size_t)N * 128 * 4);
    u16*   Hbf       = (u16*)  alloc((size_t)N * 128 * 2);
    float* Kmat      = (float*)alloc((size_t)N * 128 * 4);
    float* Hc        = (float*)alloc((size_t)N * 256 * 4);
    u16*   Hcbf      = (u16*)  alloc((size_t)N * 256 * 2);
    float* P         = (float*)alloc((size_t)N * 256 * 4);
    float* scores    = (float*)alloc((size_t)10 * N * 4);
    float* betas     = (float*)alloc((size_t)10 * N * 4);
    float* embs      = (float*)alloc((size_t)10 * 256 * 4);
    float* gammas    = (float*)alloc(64);
    (void)ws_size; (void)in_sizes; (void)n_in; (void)out_size;

    // weight prep (fp32 -> bf16)
    cast_bf16_kernel<<<(128 * 800 + 255) / 256, 256, 0, stream>>>(fe2_w, fe2wbf, 128 * 800);
    cast_bf16_kernel<<<(128 * 128 + 255) / 256, 256, 0, stream>>>(nb_w, nbwbf, 128 * 128);
    cast_bf16_kernel<<<(256 * 256 + 255) / 256, 256, 0, stream>>>(proto_w, protowbf, 256 * 256);
    prep_conv2w_kernel<<<(64 * 512 + 255) / 256, 256, 0, stream>>>(conv2_w, conv2wbf);

    // feature extractor
    conv1_pool_kernel<<<N, 256, 0, stream>>>(x, conv1_w, conv1_b, pooled1);
    conv2_pool_wmma_kernel<<<N / 2, 256, 0, stream>>>(pooled1, conv2wbf, conv2_b,
                                                      pooled2f, pooled2bf);
    gemm_bf16_kernel<128, 0><<<N / 128, 256, 0, stream>>>(pooled2bf, fe2wbf, fe2_b,
                                                          H, Hbf, 800);
    // neighbor keys: Kmat = tanh(H @ nb_w^T + nb_b)  (keys depend only on neighbor row)
    gemm_bf16_kernel<128, 1><<<N / 128, 256, 0, stream>>>(Hbf, nbwbf, nb_b,
                                                          Kmat, (u16*)nullptr, 128);
    neighbor_kernel<<<N / 256, 256, 0, stream>>>(H, Kmat, Hc, Hcbf);
    // prototype projection: P = tanh(Hc @ proto_w^T + proto_b)
    gemm_bf16_kernel<256, 1><<<N / 128, 256, 0, stream>>>(Hcbf, protowbf, proto_b,
                                                          P, (u16*)nullptr, 256);
    scores_kernel<<<N / 256, 256, 0, stream>>>(P, tmpl, scores);
    softmax_embs_kernel<<<10, 256, 0, stream>>>(scores, Hc, betas, embs);
    global_attn_kernel<<<1, 256, 0, stream>>>(embs, glob_w1, glob_b1, glob_w2, glob_b2,
                                              cls_w, cls_b, gammas, out);
    attnmap_kernel<<<N / 256, 256, 0, stream>>>(gammas, betas, out);
}
